// SchNetHamiltonian_11733850653019
// MI455X (gfx1250) — matrix operand order repaired and checked by reading, not compiled, not run on hardware
//
#include <hip/hip_runtime.h>
#include <hip/hip_bf16.h>

#define DEVINL __device__ __forceinline__

typedef __attribute__((ext_vector_type(16))) __bf16 v16bf;
typedef __attribute__((ext_vector_type(8)))  float  v8f;
typedef __attribute__((ext_vector_type(4)))  unsigned int u32x4;
typedef __attribute__((ext_vector_type(8)))  int i32x8;
typedef __attribute__((ext_vector_type(4)))  int i32x4;

// shifted softplus via fast hardware transcendentals (bf16-level accuracy)
DEVINL float sspf(float x) {
    float t = __expf(-fabsf(x));
    return fmaxf(x, 0.0f) + __logf(1.0f + t) - 0.69314718055994531f;
}

// ---- WMMA fragment loaders (bf16, 16x16x32, wave32) -------------------------
// A (16x32, MxK) from row-major [16][ld]: one contiguous 32B run (k = 16*half + j)
DEVINL v16bf load_a_frag(const __bf16* base, int ld, int k0, int lane) {
    int row = lane & 15, half = lane >> 4;
    const __bf16* p = base + row * ld + k0 + half * 16;
    v16bf a;
#pragma unroll
    for (int j = 0; j < 8; ++j) a[j] = p[j];
#pragma unroll
    for (int j = 0; j < 8; ++j) a[8 + j] = p[8 + j];
    return a;
}

// B fragments stored PRE-SWIZZLED fragment-major:
//   idx = (((k>>5)*8 + (n>>4))*32 + ((k>>4)&1)*16 + (n&15))*16 + (k&15)
// -> each lane's 16 values for tile (kt,nt) contiguous (32B run).
DEVINL v16bf load_b_frag_sw(const __bf16* base, int kt, int nt, int lane) {
    const __bf16* p = base + ((size_t)(kt * 8 + nt) * 32 + lane) * 16;
    v16bf b;
#pragma unroll
    for (int j = 0; j < 16; ++j) b[j] = p[j];
    return b;
}

DEVINL v8f wmma_bf16(v16bf a, v16bf b, v8f c) {
    return __builtin_amdgcn_wmma_f32_16x16x32_bf16(false, a, false, b, (short)0, c, false, false);
}

DEVINL int swz_idx(int k, int n) {  // NT = 8 column tiles (N = 128)
    return (((k >> 5) * 8 + (n >> 4)) * 32 + ((k >> 4) & 1) * 16 + (n & 15)) * 16 + (k & 15);
}

// ---- Tensor Data Mover: contiguous global -> LDS DMA (1-D descriptor) -------
// D# built per CDNA5 ISA 8.3/8.4: count=1, type=2, data_size=2B,
// tensor_dim0 = tile_dim0 = elems, tensor_dim1 = tile_dim1 = 1.
// This toolchain exposes the 6-arg builtin: (g0, g1, g2, g3, g_extra, cpol).
#if defined(__has_builtin)
#if __has_builtin(__builtin_amdgcn_tensor_load_to_lds) && __has_builtin(__builtin_amdgcn_s_wait_tensorcnt)
#define USE_TDM 1
#endif
#endif

#ifdef USE_TDM
DEVINL void tdm_load_lds(unsigned lds_off, const void* src, unsigned elems2B) {
    unsigned long long ga = (unsigned long long)(uintptr_t)src;
    u32x4 g0;
    g0[0] = 1u;                                               // count=1 (valid), user mode
    g0[1] = lds_off;                                          // lds_addr (bytes)
    g0[2] = (unsigned)(ga & 0xffffffffu);                     // global_addr[31:0]
    g0[3] = (unsigned)((ga >> 32) & 0x1ffffffu) | (2u << 30); // global_addr[56:32] | type=2
    i32x8 g1;
    g1[0] = (1 << 16);                                        // wg_mask=0, data_size=1 (2B)
    g1[1] = (int)((elems2B & 0xffffu) << 16);                 // tensor_dim0[15:0]
    g1[2] = (int)((elems2B >> 16) & 0xffffu) | (1 << 16);     // tensor_dim0[31:16], tensor_dim1=1
    g1[3] = (int)((elems2B & 0xffffu) << 16);                 // tile_dim0
    g1[4] = 1;                                                // tile_dim1=1, tile_dim2=0
    g1[5] = (int)elems2B;                                     // tensor_dim0_stride[31:0]
    g1[6] = 0;
    g1[7] = 0;
    i32x4 z4 = (i32x4)0;
    i32x8 z8 = (i32x8)0;
    __builtin_amdgcn_tensor_load_to_lds(g0, g1, z4, z4, z8, 0);
}
#define TDM_WAIT() __builtin_amdgcn_s_wait_tensorcnt(0)
#endif

// ---- utility kernels --------------------------------------------------------
__global__ void zero_f32(float* p, int n) {
    int stride = gridDim.x * blockDim.x;
    for (int i = blockIdx.x * blockDim.x + threadIdx.x; i < n; i += stride) p[i] = 0.0f;
}

__global__ void embed_kernel(const float* emb, const int* z, float* h, int total) {
    int i = blockIdx.x * blockDim.x + threadIdx.x;
    if (i < total) h[i] = emb[z[i >> 7] * 128 + (i & 127)];
}

// convert weights to bf16 in swizzled fragment-major order; pad mlp_w1 K 50->64
__global__ void prep_weights(const float* w1, const float* w2, const float* l1,
                             const float* l2, const float* ln,
                             __bf16* w1b, __bf16* w2b, __bf16* l1b, __bf16* l2b, __bf16* lnb) {
    int stride = gridDim.x * blockDim.x;
    for (int i = blockIdx.x * blockDim.x + threadIdx.x; i < 6 * 64 * 128; i += stride) {
        int l = i / (64 * 128);
        int k = (i / 128) & 63;
        int n = i & 127;
        float v = (k < 50) ? w1[(l * 50 + k) * 128 + n] : 0.0f;
        w1b[l * 64 * 128 + swz_idx(k, n)] = (__bf16)v;
    }
    for (int i = blockIdx.x * blockDim.x + threadIdx.x; i < 6 * 128 * 128; i += stride) {
        int l = i / (128 * 128);
        int k = (i / 128) & 127;
        int n = i & 127;
        int o = l * 128 * 128 + swz_idx(k, n);
        w2b[o] = (__bf16)w2[i];
        l1b[o] = (__bf16)l1[i];
        l2b[o] = (__bf16)l2[i];
        lnb[o] = (__bf16)ln[i];
    }
}

DEVINL void copy16(void* dst, const void* src, int elems_u4, int tid, int nthreads) {
    const uint4* s = (const uint4*)src;
    uint4* d = (uint4*)dst;
    for (int i = tid; i < elems_u4; i += nthreads) d[i] = s[i];
}

// ---- node-side GEMM: C[N,128] = act(A[N,128] @ B[128,128] + bias) (opt +=) --
// 8 waves/block; block owns 64 rows; wave w owns column tile w, loops 4 row tiles.
template <int ACT, int RES>
__global__ __launch_bounds__(256) void node_gemm(const float* __restrict__ A,
                                                 const __bf16* __restrict__ Bw,
                                                 const float* __restrict__ bias,
                                                 float* __restrict__ C) {
    extern __shared__ char smem[];
    __bf16* As = (__bf16*)smem;        // 64*128   (LDS offset 0)
    __bf16* Bs = As + 64 * 128;        // 128*128  (LDS offset 16384, swizzled)
    int tid = threadIdx.x, w = tid >> 5, lane = tid & 31;
    int row0 = blockIdx.x * 64;

#ifdef USE_TDM
    if (w == 0) tdm_load_lds(64 * 128 * 2, Bw, 128 * 128);   // async DMA of B
#else
    copy16(Bs, Bw, 128 * 128 * 2 / 16, tid, 256);
#endif
    for (int i = tid; i < 64 * 128; i += 256) As[i] = (__bf16)A[(size_t)row0 * 128 + i];
#ifdef USE_TDM
    if (w == 0) TDM_WAIT();
#endif
    __syncthreads();

    int col = w * 16 + (lane & 15);
    int half = lane >> 4;
    float bv = bias ? bias[col] : 0.0f;
#pragma unroll
    for (int rt = 0; rt < 4; ++rt) {
        v8f acc = {};
#pragma unroll
        for (int kt = 0; kt < 4; ++kt) {
            v16bf a = load_a_frag(As + rt * 16 * 128, 128, kt * 32, lane);
            v16bf b = load_b_frag_sw(Bs, kt, w, lane);
            acc = wmma_bf16(a, b, acc);
        }
#pragma unroll
        for (int r = 0; r < 8; ++r) {
            float x = acc[r] + bv;
            if (ACT) x = sspf(x);
            float* cp = &C[(size_t)(row0 + rt * 16 + r + 8 * half) * 128 + col];
            if (RES) *cp += x; else *cp = x;
        }
    }
}

// ---- fused edge kernel ------------------------------------------------------
// Per wave: 16 edges. rbf[16,64] -> GEMM1(+ssp) -> T[16,128] -> GEMM2 -> *cfac
// -> *xl[src] -> atomic scatter to agg[dst].
__global__ __launch_bounds__(256) void edge_kernel(const float* __restrict__ pos,
                                                   const int* __restrict__ esrc,
                                                   const int* __restrict__ edst,
                                                   const __bf16* __restrict__ w1b,  // [64,128] swz
                                                   const float* __restrict__ b1,
                                                   const __bf16* __restrict__ w2b,  // [128,128] swz
                                                   const float* __restrict__ b2,
                                                   const float* __restrict__ xl,
                                                   float* __restrict__ agg) {
    extern __shared__ char smem[];
    __bf16* w1s  = (__bf16*)smem;            // 64*128   (LDS offset 0, swizzled)
    __bf16* w2s  = w1s + 64 * 128;           // 128*128  (LDS offset 16384, swizzled)
    __bf16* rbfs = w2s + 128 * 128;          // 8 * 16*64
    __bf16* Ts   = rbfs + 8 * 16 * 64;       // 8 * 16*128
    float*  cfs  = (float*)(Ts + 8 * 16 * 128);  // 8*16
    int*    srcs = (int*)(cfs + 8 * 16);         // 8*16
    int*    dsts = srcs + 8 * 16;                // 8*16

    int tid = threadIdx.x, w = tid >> 5, lane = tid & 31;
    int row = lane & 15, half = lane >> 4;

#ifdef USE_TDM
    if (w == 0) {                                     // wave 0 issues async weight DMA
        tdm_load_lds(0, w1b, 64 * 128);
        tdm_load_lds(64 * 128 * 2, w2b, 128 * 128);
    }
#else
    copy16(w1s, w1b, 64 * 128 * 2 / 16, tid, 256);
    copy16(w2s, w2b, 128 * 128 * 2 / 16, tid, 256);
#endif

    // --- distances, gaussian smearing, cosine cutoff (overlaps the DMA) ---
    int e = (blockIdx.x * 8 + w) * 16 + row;
    int s = esrc[e], d = edst[e];
    float dx = pos[s * 3 + 0] - pos[d * 3 + 0];
    float dy = pos[s * 3 + 1] - pos[d * 3 + 1];
    float dz = pos[s * 3 + 2] - pos[d * 3 + 2];
    float r = sqrtf(dx * dx + dy * dy + dz * dz + 1e-12f);
    if (half == 0) {
        srcs[w * 16 + row] = s;
        dsts[w * 16 + row] = d;
        float c = 0.5f * (__cosf(r * (3.14159265358979f / 5.0f)) + 1.0f);
        cfs[w * 16 + row] = (r < 5.0f) ? c : 0.0f;
    }
    const float delta = 5.0f / 49.0f;
    const float coeff = -0.5f / (delta * delta);
    __bf16* rb = rbfs + w * 16 * 64;
#pragma unroll
    for (int gi = 0; gi < 32; ++gi) {
        int g = half * 32 + gi;
        float t = r - (float)g * delta;
        float v = (g < 50) ? __expf(coeff * t * t) : 0.0f;
        rb[row * 64 + g] = (__bf16)v;
    }
#ifdef USE_TDM
    if (w == 0) TDM_WAIT();
#endif
    __syncthreads();

    // --- GEMM1: [16,64] @ [64,128], + b1, ssp, -> T bf16 ---
    __bf16* T = Ts + w * 16 * 128;
#pragma unroll
    for (int nt = 0; nt < 8; ++nt) {
        v8f acc = {};
#pragma unroll
        for (int kt = 0; kt < 2; ++kt) {
            v16bf a = load_a_frag(rb, 64, kt * 32, lane);
            v16bf b = load_b_frag_sw(w1s, kt, nt, lane);
            acc = wmma_bf16(a, b, acc);
        }
        int col = nt * 16 + row;
        float bv = b1[col];
#pragma unroll
        for (int r2 = 0; r2 < 8; ++r2)
            T[(r2 + 8 * half) * 128 + col] = (__bf16)sspf(acc[r2] + bv);
    }
    __syncthreads();

    // --- GEMM2: [16,128] @ [128,128], + b2, * cfac, * xl[src], scatter ---
#pragma unroll
    for (int nt = 0; nt < 8; ++nt) {
        v8f acc = {};
#pragma unroll
        for (int kt = 0; kt < 4; ++kt) {
            v16bf a = load_a_frag(T, 128, kt * 32, lane);
            v16bf b = load_b_frag_sw(w2s, kt, nt, lane);
            acc = wmma_bf16(a, b, acc);
        }
        int col = nt * 16 + row;
        float bv = b2[col];
#pragma unroll
        for (int r2 = 0; r2 < 8; ++r2) {
            int rr = r2 + 8 * half;
            float Wv = (acc[r2] + bv) * cfs[w * 16 + rr];
            float v  = Wv * xl[(size_t)srcs[w * 16 + rr] * 128 + col];
            __hip_atomic_fetch_add(&agg[(size_t)dsts[w * 16 + rr] * 128 + col], v,
                                   __ATOMIC_RELAXED, __HIP_MEMORY_SCOPE_AGENT);
        }
    }
}

// ---- output head: o = ssp(h@W1 + b1) @ W2 + b2, sum per molecule ------------
__global__ void head_kernel(const float* __restrict__ h, const float* __restrict__ w1,
                            const float* __restrict__ b1, const float* __restrict__ w2,
                            const float* __restrict__ b2, const int* __restrict__ batch,
                            float* __restrict__ out, int N) {
    int n = blockIdx.x * blockDim.x + threadIdx.x;
    if (n >= N) return;
    float hr[128];
#pragma unroll
    for (int c = 0; c < 128; ++c) hr[c] = h[(size_t)n * 128 + c];
    float o = b2[0];
    for (int j = 0; j < 64; ++j) {
        float s = b1[j];
#pragma unroll
        for (int c = 0; c < 128; ++c) s = fmaf(hr[c], w1[c * 64 + j], s);
        o = fmaf(sspf(s), w2[j], o);
    }
    __hip_atomic_fetch_add(&out[batch[n]], o, __ATOMIC_RELAXED, __HIP_MEMORY_SCOPE_AGENT);
}

// ---- host launcher ----------------------------------------------------------
extern "C" void kernel_launch(void* const* d_in, const int* in_sizes, int n_in,
                              void* d_out, int out_size, void* d_ws, size_t ws_size,
                              hipStream_t stream) {
    (void)n_in; (void)ws_size;
    const int*   z      = (const int*)d_in[0];
    const float* pos    = (const float*)d_in[1];
    const int*   batch  = (const int*)d_in[2];
    const int*   esrc   = (const int*)d_in[3];
    const int*   edst   = (const int*)d_in[4];
    const float* emb    = (const float*)d_in[5];
    const float* mlp_w1 = (const float*)d_in[6];
    const float* mlp_b1 = (const float*)d_in[7];
    const float* mlp_w2 = (const float*)d_in[8];
    const float* mlp_b2 = (const float*)d_in[9];
    const float* lin1_w = (const float*)d_in[10];
    const float* lin2_w = (const float*)d_in[11];
    const float* lin2_b = (const float*)d_in[12];
    const float* lin_w  = (const float*)d_in[13];
    const float* lin_b  = (const float*)d_in[14];
    const float* out1_w = (const float*)d_in[15];
    const float* out1_b = (const float*)d_in[16];
    const float* out2_w = (const float*)d_in[17];
    const float* out2_b = (const float*)d_in[18];

    const int N = in_sizes[0];      // 24576
    const int E = in_sizes[3];      // 786432
    const int Mmol = out_size;      // 512
    const int L = 6;

    // workspace partition (256B aligned)
    char* wsp = (char*)d_ws;
    auto alloc = [&](size_t bytes) { char* p = wsp; wsp += (bytes + 255) & ~(size_t)255; return p; };
    __bf16* w1b = (__bf16*)alloc((size_t)L * 64 * 128 * 2);
    __bf16* w2b = (__bf16*)alloc((size_t)L * 128 * 128 * 2);
    __bf16* l1b = (__bf16*)alloc((size_t)L * 128 * 128 * 2);
    __bf16* l2b = (__bf16*)alloc((size_t)L * 128 * 128 * 2);
    __bf16* lnb = (__bf16*)alloc((size_t)L * 128 * 128 * 2);
    float* h    = (float*)alloc((size_t)N * 128 * 4);
    float* xl   = (float*)alloc((size_t)N * 128 * 4);
    float* agg  = (float*)alloc((size_t)N * 128 * 4);
    float* tbuf = (float*)alloc((size_t)N * 128 * 4);

    const size_t smemG = (size_t)(64 * 128 + 128 * 128) * 2;                       // 48 KB
    const size_t smemE = (size_t)(64 * 128 + 128 * 128 + 8 * 16 * 64 + 8 * 16 * 128) * 2
                         + 8 * 16 * 4 + 2 * 8 * 16 * 4;                            // ~97.5 KB
    (void)hipFuncSetAttribute(reinterpret_cast<const void*>(&edge_kernel),
                              hipFuncAttributeMaxDynamicSharedMemorySize, (int)smemE);

    prep_weights<<<128, 256, 0, stream>>>(mlp_w1, mlp_w2, lin1_w, lin2_w, lin_w,
                                          w1b, w2b, l1b, l2b, lnb);
    embed_kernel<<<(N * 128 + 255) / 256, 256, 0, stream>>>(emb, z, h, N * 128);
    zero_f32<<<4, 128, 0, stream>>>((float*)d_out, Mmol);

    for (int l = 0; l < L; ++l) {
        node_gemm<0, 0><<<N / 64, 256, smemG, stream>>>(h, l1b + (size_t)l * 128 * 128, nullptr, xl);
        zero_f32<<<2048, 256, 0, stream>>>(agg, N * 128);
        edge_kernel<<<E / 128, 256, smemE, stream>>>(pos, esrc, edst,
                                                     w1b + (size_t)l * 64 * 128, mlp_b1 + l * 128,
                                                     w2b + (size_t)l * 128 * 128, mlp_b2 + l * 128,
                                                     xl, agg);
        node_gemm<1, 0><<<N / 64, 256, smemG, stream>>>(agg, l2b + (size_t)l * 128 * 128,
                                                        lin2_b + l * 128, tbuf);
        node_gemm<0, 1><<<N / 64, 256, smemG, stream>>>(tbuf, lnb + (size_t)l * 128 * 128,
                                                        lin_b + l * 128, h);
    }
    head_kernel<<<N / 256, 256, 0, stream>>>(h, out1_w, out1_b, out2_w, out2_b, batch,
                                             (float*)d_out, N);
}
